// SegBlock_50955492000310
// MI455X (gfx1250) — matrix operand order repaired and checked by reading, not compiled, not run on hardware
//
#include <hip/hip_runtime.h>
#include <hip/hip_bf16.h>

typedef __bf16 bf16;
typedef __attribute__((ext_vector_type(16))) __bf16 v16bf;
typedef __attribute__((ext_vector_type(8)))  __bf16 bf16x8;
typedef __attribute__((ext_vector_type(8)))  float  v8f;

#define BATCH 4
#define HDIM  48
#define HW    2304
#define NQTOK 8
#define NTOK  2312
#define NPAD  2336
#define CDIM  256
#define NHEAD 8
#define FFN   1024
#define ROWS  (BATCH*NPAD)

union Frag { v16bf v; bf16x8 h[2]; };

__device__ __forceinline__ bf16x8 bzero8() {
  bf16x8 z;
#pragma unroll
  for (int i = 0; i < 8; ++i) z[i] = (bf16)0.0f;
  return z;
}
__device__ __forceinline__ v8f fzero8() {
  v8f z;
#pragma unroll
  for (int i = 0; i < 8; ++i) z[i] = 0.0f;
  return z;
}

// ---------------------------------------------------------------------------
// Elementwise / setup kernels
// ---------------------------------------------------------------------------

// rows 0..7 per batch = queries; rows NTOK..NPAD-1 = zero padding
__global__ __launch_bounds__(256) void initxq_kernel(
    const float* __restrict__ queries, float* __restrict__ XQ)
{
  int c = threadIdx.x;
  int r = blockIdx.x;           // 0 .. BATCH*32-1
  int b = r >> 5, k = r & 31;
  if (k < NQTOK)
    XQ[((size_t)b*NPAD + k)*CDIM + c] = queries[((size_t)b*NQTOK + k)*CDIM + c];
  else
    XQ[((size_t)b*NPAD + NTOK + (k - NQTOK))*CDIM + c] = 0.0f;
}

// xq[8+pix] = x + dwconv3x3(x)
__global__ __launch_bounds__(256) void posconv_kernel(
    const float* __restrict__ x, const float* __restrict__ pw,
    const float* __restrict__ pb, float* __restrict__ XQ)
{
  int c = threadIdx.x;
  int pix = blockIdx.x % HW;
  int b   = blockIdx.x / HW;
  int i = pix / HDIM, j = pix % HDIM;
  const float* xb = x + (size_t)b*HW*CDIM;
  float acc = pb[c];
#pragma unroll
  for (int ki = -1; ki <= 1; ++ki)
#pragma unroll
    for (int kj = -1; kj <= 1; ++kj) {
      int ii = i + ki, jj = j + kj;
      if (ii >= 0 && ii < HDIM && jj >= 0 && jj < HDIM)
        acc += xb[((size_t)ii*HDIM + jj)*CDIM + c] *
               pw[((ki+1)*3 + (kj+1))*CDIM + c];
    }
  XQ[((size_t)b*NPAD + NQTOK + pix)*CDIM + c] = xb[(size_t)pix*CDIM + c] + acc;
}

// p = softmax(xf . queries / tau), stored bf16 (b, NPAD rows, 8)
__global__ __launch_bounds__(32) void pmask_kernel(
    const float* __restrict__ XQ, const float* __restrict__ queries,
    const int* __restrict__ epoch, bf16* __restrict__ P)
{
  int lane = threadIdx.x & 31;
  int pix = blockIdx.x % HW;
  int b   = blockIdx.x / HW;
  const float* xr = XQ + ((size_t)b*NPAD + NQTOK + pix)*CDIM;
  float acc[8];
#pragma unroll
  for (int q = 0; q < 8; ++q) acc[q] = 0.0f;
  for (int k = 0; k < 8; ++k) {
    float xv = xr[lane*8 + k];
#pragma unroll
    for (int q = 0; q < 8; ++q)
      acc[q] += xv * queries[((size_t)b*NQTOK + q)*CDIM + lane*8 + k];
  }
#pragma unroll
  for (int q = 0; q < 8; ++q)
    for (int d = 1; d < 32; d <<= 1)
      acc[q] += __shfl_xor(acc[q], d, 32);
  float tau = 2.0f * powf(0.05f, (float)epoch[0] / 200.0f);
  float mx = acc[0];
#pragma unroll
  for (int q = 1; q < 8; ++q) mx = fmaxf(mx, acc[q]);
  float e[8], s = 0.0f;
#pragma unroll
  for (int q = 0; q < 8; ++q) { e[q] = __expf((acc[q] - mx) / tau); s += e[q]; }
  if (lane < 8)
    P[((size_t)b*NPAD + pix)*8 + lane] = (bf16)(e[lane] / s);
}

__global__ __launch_bounds__(256) void cvt_kernel(
    const float* __restrict__ src, bf16* __restrict__ dst, int n)
{
  int i = blockIdx.x * 256 + threadIdx.x;
  if (i < n) dst[i] = (bf16)src[i];
}

// LayerNorm over C=256 -> bf16 (zero for padded rows)
__global__ __launch_bounds__(256) void ln_kernel(
    const float* __restrict__ X, const float* __restrict__ g,
    const float* __restrict__ bb, bf16* __restrict__ Y)
{
  __shared__ float red[256];
  int row = blockIdx.x;          // b*NPAD + n
  int n = row % NPAD;
  int c = threadIdx.x;
  if (n >= NTOK) { Y[(size_t)row*CDIM + c] = (bf16)0.0f; return; }
  float x = X[(size_t)row*CDIM + c];
  red[c] = x; __syncthreads();
  for (int s = 128; s > 0; s >>= 1) { if (c < s) red[c] += red[c+s]; __syncthreads(); }
  float mean = red[0] * (1.0f/256.0f); __syncthreads();
  float dx = x - mean;
  red[c] = dx*dx; __syncthreads();
  for (int s = 128; s > 0; s >>= 1) { if (c < s) red[c] += red[c+s]; __syncthreads(); }
  float var = red[0] * (1.0f/256.0f);
  Y[(size_t)row*CDIM + c] = (bf16)(dx * rsqrtf(var + 1e-6f) * g[c] + bb[c]);
}

// olepe = o (+ dwconv5x5(v) for spatial rows), bf16; zero padding
__global__ __launch_bounds__(256) void lepe_kernel(
    const float* __restrict__ O_, const float* __restrict__ V,
    const float* __restrict__ lw, const float* __restrict__ lb,
    bf16* __restrict__ OL)
{
  int c = threadIdx.x;
  int row = blockIdx.x;          // b*NPAD + n
  int b = row / NPAD, n = row % NPAD;
  size_t idx = (size_t)row*CDIM + c;
  if (n >= NTOK) { OL[idx] = (bf16)0.0f; return; }
  float val = O_[idx];
  if (n >= NQTOK) {
    int pix = n - NQTOK;
    int i = pix / HDIM, j = pix % HDIM;
    float acc = lb[c];
#pragma unroll
    for (int ki = -2; ki <= 2; ++ki)
#pragma unroll
      for (int kj = -2; kj <= 2; ++kj) {
        int ii = i + ki, jj = j + kj;
        if (ii >= 0 && ii < HDIM && jj >= 0 && jj < HDIM)
          acc += V[((size_t)b*NPAD + NQTOK + ii*HDIM + jj)*CDIM + c] *
                 lw[((ki+2)*5 + (kj+2))*CDIM + c];
      }
    val += acc;
  }
  OL[idx] = (bf16)val;
}

__global__ __launch_bounds__(256) void output_kernel(
    const float* __restrict__ XQ, float* __restrict__ out)
{
  int c = threadIdx.x;
  int r = blockIdx.x;            // BATCH*(HW+8)
  int b = r / (HW + NQTOK), k = r % (HW + NQTOK);
  if (k < HW)
    out[((size_t)b*HW + k)*CDIM + c] = XQ[((size_t)b*NPAD + NQTOK + k)*CDIM + c];
  else
    out[(size_t)BATCH*HW*CDIM + ((size_t)b*NQTOK + (k - HW))*CDIM + c] =
        XQ[((size_t)b*NPAD + (k - HW))*CDIM + c];
}

// ---------------------------------------------------------------------------
// WMMA GEMM with async-LDS weight staging (CDNA5 GLOBAL_LOAD_ASYNC_TO_LDS).
// 256-thread block = 8 waves; block computes 128 rows x 64 cols.
// The 64-col W panel is staged per 32-k chunk into double-buffered LDS via
// global_load_async_to_lds_b128 (one 16B transfer per thread per stage),
// synchronized with s_wait_asynccnt; all 8 waves consume it via ds loads.
// epi: 1=store bf16; 2=scale, store bf16 + transposed kT; 3=store f32;
//      4=accumulate into f32; 5=GELU -> bf16
// ---------------------------------------------------------------------------
__global__ __launch_bounds__(256) void wmma_gemm_kernel(
    const bf16* __restrict__ A, const bf16* __restrict__ W,
    const float* __restrict__ bias, int K, int Nout, int epi,
    float* __restrict__ fout, bf16* __restrict__ bout,
    bf16* __restrict__ kT, float kscale)
{
  __shared__ bf16 wt[2][64*32];          // [buf][panel_col][k] : 4KB per buffer
  const int tid  = threadIdx.x;
  const int lane = tid & 31;
  const int wave = tid >> 5;
  const int rsel = lane & 15;
  const int half = (lane >> 4) & 1;
  const int row0 = blockIdx.y * 128 + wave * 16;
  const int col0 = blockIdx.x * 64;

  // async-copy mapping: each thread moves 16B (8 bf16) of the W panel
  const int crow = tid >> 2;             // 0..63 : panel column (row of W)
  const int ckof = (tid & 3) * 8;        // 0,8,16,24 within 32-k chunk
  const unsigned lds0 = (unsigned)(uintptr_t)&wt[0][crow*32 + ckof];
  const unsigned lds1 = (unsigned)(uintptr_t)&wt[1][crow*32 + ckof];
  const bf16* gsrc = W + (size_t)(col0 + crow) * K + ckof;

  // stage 0 -> buffer 0
  asm volatile("global_load_async_to_lds_b128 %0, %1, off"
               :: "v"(lds0), "v"((unsigned long long)(uintptr_t)gsrc)
               : "memory");

  v8f acc[4];
#pragma unroll
  for (int t = 0; t < 4; ++t) acc[t] = fzero8();

  const bf16* arow = A + (size_t)(row0 + rsel) * K;
  const int nsteps = K >> 5;

  for (int s = 0; s < nsteps; ++s) {
    const int k0 = s << 5;
    if (s + 1 < nsteps) {
      // issue next stage into the other buffer, then wait for current stage
      unsigned dst = (s & 1) ? lds0 : lds1;
      asm volatile("global_load_async_to_lds_b128 %0, %1, off"
                   :: "v"(dst),
                      "v"((unsigned long long)(uintptr_t)(gsrc + k0 + 32))
                   : "memory");
      asm volatile("s_wait_asynccnt 1" ::: "memory");
    } else {
      asm volatile("s_wait_asynccnt 0" ::: "memory");
    }
    __syncthreads();                     // all waves' panel slices arrived

    Frag af;
    af.h[0] = *(const bf16x8*)(arow + k0 + half*8);
    af.h[1] = *(const bf16x8*)(arow + k0 + half*8 + 16);
    const bf16* wb = wt[s & 1];
#pragma unroll
    for (int t = 0; t < 4; ++t) {
      const bf16* wrow = wb + (t*16 + rsel)*32 + half*8;
      Frag bfg;
      bfg.h[0] = *(const bf16x8*)(wrow);
      bfg.h[1] = *(const bf16x8*)(wrow + 16);
      acc[t] = __builtin_amdgcn_wmma_f32_16x16x32_bf16(
          false, af.v, false, bfg.v, (short)0, acc[t], false, false);
    }
    __syncthreads();                     // done reading before next overwrite
  }

  const int rbase = row0 + half*8;
#pragma unroll
  for (int t = 0; t < 4; ++t) {
    int col = col0 + t*16 + rsel;
    float bv = bias[col];
#pragma unroll
    for (int e = 0; e < 8; ++e) {
      int row = rbase + e;
      float val = acc[t][e] + bv;
      if (epi == 1) {
        bout[(size_t)row*Nout + col] = (bf16)val;
      } else if (epi == 2) {
        val *= kscale;
        bout[(size_t)row*Nout + col] = (bf16)val;
        int b = row / NPAD, n = row % NPAD;
        int hh = col >> 5, d = col & 31;
        kT[(((size_t)b*NHEAD + hh)*32 + d)*NPAD + n] = (bf16)val;
      } else if (epi == 3) {
        fout[(size_t)row*Nout + col] = val;
      } else if (epi == 4) {
        fout[(size_t)row*Nout + col] += val;
      } else {
        float gl = 0.5f * val * (1.0f + erff(val * 0.70710678118f));
        bout[(size_t)row*Nout + col] = (bf16)gl;
      }
    }
  }
}

// ---------------------------------------------------------------------------
// Fused flash attention with WMMA-computed region mask.
// One wave per (batch, head, 16-row tile). Per 32-column step:
//   2x QK^T wmma + 2x mask wmma (p . p^T, K padded 8->32) + 2x P.V wmma
// vh = scaled k (faithful to reference). Online softmax in fp32.
// ---------------------------------------------------------------------------
__global__ __launch_bounds__(32) void attn_kernel(
    const bf16* __restrict__ Q, const bf16* __restrict__ Kb,
    const bf16* __restrict__ KT, const bf16* __restrict__ P,
    float* __restrict__ O)
{
  __shared__ bf16 pt[16*32];
  const int lane = threadIdx.x & 31;
  const int rsel = lane & 15;
  const int half = lane >> 4;
  const int mt = blockIdx.x;     // row tile (NPAD/16)
  const int h  = blockIdx.y;
  const int b  = blockIdx.z;
  const int n0 = mt * 16;

  // Q A-fragment (16 rows x K=32 head dims)
  Frag qa;
  {
    const bf16* p = Q + ((size_t)b*NPAD + n0 + rsel)*CDIM + h*32 + half*16;
    qa.h[0] = *(const bf16x8*)p;
    qa.h[1] = *(const bf16x8*)(p + 8);
  }
  // mask A-fragment: p rows for this row tile, K=8 valid, rest zero
  Frag pa;
  {
    int pr = n0 + rsel - NQTOK; if (pr < 0) pr = 0;
    pa.h[0] = (half == 0) ? *(const bf16x8*)(P + ((size_t)b*NPAD + pr)*8)
                          : bzero8();
    pa.h[1] = bzero8();
  }

  float mx[8], ls[8];
#pragma unroll
  for (int e = 0; e < 8; ++e) { mx[e] = -3.0e38f; ls[e] = 0.0f; }
  v8f o0 = fzero8(), o1 = fzero8();
  const v8f cz = fzero8();

  for (int m0 = 0; m0 < NPAD; m0 += 32) {
    // K B-fragments (K = head dim, contiguous): columns m0..m0+15 / +16..+31
    Frag kb0, kb1;
    {
      const bf16* kp = Kb + ((size_t)b*NPAD + m0 + rsel)*CDIM + h*32 + half*16;
      kb0.h[0] = *(const bf16x8*)kp;
      kb0.h[1] = *(const bf16x8*)(kp + 8);
      const bf16* kq = kp + (size_t)16*CDIM;
      kb1.h[0] = *(const bf16x8*)kq;
      kb1.h[1] = *(const bf16x8*)(kq + 8);
    }
    v8f s0 = __builtin_amdgcn_wmma_f32_16x16x32_bf16(false, qa.v, false, kb0.v,
                                                     (short)0, cz, false, false);
    v8f s1 = __builtin_amdgcn_wmma_f32_16x16x32_bf16(false, qa.v, false, kb1.v,
                                                     (short)0, cz, false, false);
    // mask B-fragments: p rows for columns, K=8 valid
    Frag mb0, mb1;
    {
      int pm0 = m0 + rsel - NQTOK;      if (pm0 < 0) pm0 = 0;
      int pm1 = m0 + 16 + rsel - NQTOK; if (pm1 < 0) pm1 = 0;
      mb0.h[0] = (half == 0) ? *(const bf16x8*)(P + ((size_t)b*NPAD + pm0)*8)
                             : bzero8();
      mb0.h[1] = bzero8();
      mb1.h[0] = (half == 0) ? *(const bf16x8*)(P + ((size_t)b*NPAD + pm1)*8)
                             : bzero8();
      mb1.h[1] = bzero8();
    }
    v8f mk0 = __builtin_amdgcn_wmma_f32_16x16x32_bf16(false, pa.v, false, mb0.v,
                                                      (short)0, cz, false, false);
    v8f mk1 = __builtin_amdgcn_wmma_f32_16x16x32_bf16(false, pa.v, false, mb1.v,
                                                      (short)0, cz, false, false);

    const int c0 = m0 + rsel, c1 = c0 + 16;
    float p0[8], p1[8], tmax[8];
#pragma unroll
    for (int e = 0; e < 8; ++e) {
      int row = n0 + half*8 + e;
      float m0v = (row < NQTOK || c0 < NQTOK) ? 1.0f : mk0[e];
      float m1v = (row < NQTOK || c1 < NQTOK) ? 1.0f : mk1[e];
      p0[e] = (c0 < NTOK) ? s0[e]*m0v : -3.0e38f;
      p1[e] = (c1 < NTOK) ? s1[e]*m1v : -3.0e38f;
      tmax[e] = fmaxf(p0[e], p1[e]);
    }
#pragma unroll
    for (int e = 0; e < 8; ++e)
      for (int d = 1; d < 16; d <<= 1)
        tmax[e] = fmaxf(tmax[e], __shfl_xor(tmax[e], d, 32));
#pragma unroll
    for (int e = 0; e < 8; ++e) {
      float nm = fmaxf(mx[e], tmax[e]);
      float alpha = __expf(mx[e] - nm);
      mx[e] = nm;
      p0[e] = __expf(p0[e] - nm);
      p1[e] = __expf(p1[e] - nm);
      float rs = p0[e] + p1[e];
      for (int d = 1; d < 16; d <<= 1) rs += __shfl_xor(rs, d, 32);
      ls[e] = ls[e]*alpha + rs;
      o0[e] *= alpha;
      o1[e] *= alpha;
      pt[(half*8 + e)*32 + rsel]      = (bf16)p0[e];
      pt[(half*8 + e)*32 + rsel + 16] = (bf16)p1[e];
    }
    asm volatile("s_wait_dscnt 0" ::: "memory");
    // P A-fragment from LDS (C-layout -> A-layout bounce)
    Frag pf;
    {
      const bf16* lp = pt + rsel*32 + half*16;
      pf.h[0] = *(const bf16x8*)lp;
      pf.h[1] = *(const bf16x8*)(lp + 8);
    }
    // V (= scaled k) B-fragments from transposed kT: contiguous over m
    Frag v0, v1;
    {
      const bf16* vp0 = KT + (((size_t)b*NHEAD + h)*32 + rsel)*NPAD + m0 + half*16;
      v0.h[0] = *(const bf16x8*)vp0;
      v0.h[1] = *(const bf16x8*)(vp0 + 8);
      const bf16* vp1 = vp0 + (size_t)16*NPAD;
      v1.h[0] = *(const bf16x8*)vp1;
      v1.h[1] = *(const bf16x8*)(vp1 + 8);
    }
    o0 = __builtin_amdgcn_wmma_f32_16x16x32_bf16(false, pf.v, false, v0.v,
                                                 (short)0, o0, false, false);
    o1 = __builtin_amdgcn_wmma_f32_16x16x32_bf16(false, pf.v, false, v1.v,
                                                 (short)0, o1, false, false);
  }

#pragma unroll
  for (int e = 0; e < 8; ++e) {
    float inv = 1.0f / ls[e];
    int n = n0 + half*8 + e;
    float* op = O + ((size_t)b*NPAD + n)*CDIM + h*32;
    op[rsel]      = o0[e] * inv;
    op[rsel + 16] = o1[e] * inv;
  }
}

// ---------------------------------------------------------------------------
// Host launcher
// ---------------------------------------------------------------------------
extern "C" void kernel_launch(void* const* d_in, const int* in_sizes, int n_in,
                              void* d_out, int out_size, void* d_ws, size_t ws_size,
                              hipStream_t stream)
{
  (void)in_sizes; (void)n_in; (void)out_size; (void)ws_size;
  const float* x       = (const float*)d_in[0];
  const float* queries = (const float*)d_in[1];
  const int*   epoch   = (const int*)  d_in[2];
  const float* pos_w   = (const float*)d_in[3];
  const float* pos_b   = (const float*)d_in[4];
  const float* ln1_g   = (const float*)d_in[5];
  const float* ln1_b   = (const float*)d_in[6];
  const float* q_w     = (const float*)d_in[7];
  const float* q_b     = (const float*)d_in[8];
  const float* k_w     = (const float*)d_in[9];
  const float* k_b     = (const float*)d_in[10];
  const float* v_w     = (const float*)d_in[11];
  const float* v_b     = (const float*)d_in[12];
  const float* lepe_w  = (const float*)d_in[13];
  const float* lepe_b  = (const float*)d_in[14];
  const float* out_w   = (const float*)d_in[15];
  const float* out_b   = (const float*)d_in[16];
  const float* ln2_g   = (const float*)d_in[17];
  const float* ln2_b   = (const float*)d_in[18];
  const float* fc1_w   = (const float*)d_in[19];
  const float* fc1_b   = (const float*)d_in[20];
  const float* fc2_w   = (const float*)d_in[21];
  const float* fc2_b   = (const float*)d_in[22];

  char* base = (char*)d_ws;
  size_t off = 0;
  auto alloc = [&](size_t bytes) -> char* {
    off = (off + 255) & ~(size_t)255;
    char* p = base + off;
    off += bytes;
    return p;
  };
  float* XQ   = (float*)alloc((size_t)ROWS*CDIM*4);
  bf16*  Ybf  = (bf16*) alloc((size_t)ROWS*CDIM*2);
  bf16*  Qbf  = (bf16*) alloc((size_t)ROWS*CDIM*2);
  bf16*  Kbf  = (bf16*) alloc((size_t)ROWS*CDIM*2);
  bf16*  KTbf = (bf16*) alloc((size_t)ROWS*CDIM*2);
  float* Vf   = (float*)alloc((size_t)ROWS*CDIM*4);
  float* Of   = (float*)alloc((size_t)ROWS*CDIM*4);
  bf16*  OLbf = (bf16*) alloc((size_t)ROWS*CDIM*2);
  bf16*  Zbf  = (bf16*) alloc((size_t)ROWS*CDIM*2);
  bf16*  H1bf = (bf16*) alloc((size_t)ROWS*FFN*2);
  bf16*  Pbf  = (bf16*) alloc((size_t)ROWS*8*2);
  bf16*  Wq   = (bf16*) alloc((size_t)CDIM*CDIM*2);
  bf16*  Wk   = (bf16*) alloc((size_t)CDIM*CDIM*2);
  bf16*  Wv   = (bf16*) alloc((size_t)CDIM*CDIM*2);
  bf16*  Wo   = (bf16*) alloc((size_t)CDIM*CDIM*2);
  bf16*  Wf1  = (bf16*) alloc((size_t)FFN*CDIM*2);
  bf16*  Wf2  = (bf16*) alloc((size_t)CDIM*FFN*2);

  hipMemsetAsync(Pbf, 0, (size_t)ROWS*8*2, stream);

  initxq_kernel<<<BATCH*32, 256, 0, stream>>>(queries, XQ);
  posconv_kernel<<<BATCH*HW, 256, 0, stream>>>(x, pos_w, pos_b, XQ);
  pmask_kernel<<<BATCH*HW, 32, 0, stream>>>(XQ, queries, epoch, Pbf);

  const int NW = CDIM*CDIM;
  cvt_kernel<<<(NW+255)/256, 256, 0, stream>>>(q_w,   Wq,  NW);
  cvt_kernel<<<(NW+255)/256, 256, 0, stream>>>(k_w,   Wk,  NW);
  cvt_kernel<<<(NW+255)/256, 256, 0, stream>>>(v_w,   Wv,  NW);
  cvt_kernel<<<(NW+255)/256, 256, 0, stream>>>(out_w, Wo,  NW);
  const int NF = FFN*CDIM;
  cvt_kernel<<<(NF+255)/256, 256, 0, stream>>>(fc1_w, Wf1, NF);
  cvt_kernel<<<(NF+255)/256, 256, 0, stream>>>(fc2_w, Wf2, NF);

  ln_kernel<<<ROWS, 256, 0, stream>>>(XQ, ln1_g, ln1_b, Ybf);

  const int MB = ROWS/128;             // 73 row blocks (128 rows each)
  const float kscale = 0.1767766953f;  // 1/sqrt(32)
  wmma_gemm_kernel<<<dim3(CDIM/64, MB), 256, 0, stream>>>(
      Ybf, Wq, q_b, CDIM, CDIM, 1, nullptr, Qbf, nullptr, 1.0f);
  wmma_gemm_kernel<<<dim3(CDIM/64, MB), 256, 0, stream>>>(
      Ybf, Wk, k_b, CDIM, CDIM, 2, nullptr, Kbf, KTbf, kscale);
  wmma_gemm_kernel<<<dim3(CDIM/64, MB), 256, 0, stream>>>(
      Ybf, Wv, v_b, CDIM, CDIM, 3, Vf, nullptr, nullptr, 1.0f);

  attn_kernel<<<dim3(NPAD/16, NHEAD, BATCH), 32, 0, stream>>>(
      Qbf, Kbf, KTbf, Pbf, Of);

  lepe_kernel<<<ROWS, 256, 0, stream>>>(Of, Vf, lepe_w, lepe_b, OLbf);

  wmma_gemm_kernel<<<dim3(CDIM/64, MB), 256, 0, stream>>>(
      OLbf, Wo, out_b, CDIM, CDIM, 4, XQ, nullptr, nullptr, 1.0f);

  ln_kernel<<<ROWS, 256, 0, stream>>>(XQ, ln2_g, ln2_b, Zbf);

  wmma_gemm_kernel<<<dim3(FFN/64, MB), 256, 0, stream>>>(
      Zbf, Wf1, fc1_b, CDIM, FFN, 5, nullptr, H1bf, nullptr, 1.0f);
  wmma_gemm_kernel<<<dim3(CDIM/64, MB), 256, 0, stream>>>(
      H1bf, Wf2, fc2_b, FFN, CDIM, 4, XQ, nullptr, nullptr, 1.0f);

  output_kernel<<<BATCH*(HW+NQTOK), 256, 0, stream>>>(XQ, (float*)d_out);
}
